// architecture_23433341566984
// MI455X (gfx1250) — matrix operand order repaired and checked by reading, not compiled, not run on hardware
//
#include <hip/hip_runtime.h>
#include <hip/hip_bf16.h>

// ---- problem constants (match reference) ----
constexpr int Hn   = 256;   // hidden
constexpr int Pn   = 8;     // input features
constexpr int PREn = 96;    // encoder prefix
constexpr int SEQn = 240;   // total x timesteps
constexpr int TDEC = 144;   // decoder steps
constexpr int OUTn = 96;    // emitted outputs
constexpr int KT   = 9;     // k-tiles of 32 over K = 256 (h) + 8 (x) padded to 288
constexpr int NTG  = 48;    // GRU N tiles (3*256/16)
constexpr int NTM  = 16;    // MLP N tiles (256/16)

typedef __attribute__((ext_vector_type(16))) __bf16 v16bf;
typedef __attribute__((ext_vector_type(8)))  float  v8f;

union BFrag { v16bf v; uint4 q[2]; unsigned short s[16]; };

__device__ __forceinline__ unsigned short f2bf(float f) {
  union { float f; unsigned u; } a; a.f = f;
  unsigned r = a.u + 0x7FFFu + ((a.u >> 16) & 1u);   // round-to-nearest-even
  return (unsigned short)(r >> 16);
}
__device__ __forceinline__ float bf2f(unsigned short b) {
  union { float f; unsigned u; } a; a.u = ((unsigned)b) << 16;
  return a.f;
}
__device__ __forceinline__ float sigf(float x) { return 1.f / (1.f + __expf(-x)); }
__device__ __forceinline__ float tanhfast(float x) {
  float e = __expf(2.f * x);
  return (e - 1.f) / (e + 1.f);
}

// ---------------------------------------------------------------------------
// Weight prep: build pre-swizzled bf16 B-matrix fragments.
// Fragment layout (32x16 K x N bf16 B operand, wave32):
//   lane l: N = nt*16 + (l & 15), K-segment = (l >= 16) ? 16 : 0,
//   16 contiguous bf16 per lane: element e <-> K = kt*32 + kseg + e.
// Combined K axis: 0..255 = h (w_hh / w1 hidden part), 256.. = x part, pad 0.
// ---------------------------------------------------------------------------
__global__ void prep_weights(const float* __restrict__ w_ih,
                             const float* __restrict__ w_hh,
                             const float* __restrict__ w1,
                             unsigned short* __restrict__ Wg,
                             unsigned short* __restrict__ Wm) {
  const int totalG = NTG * KT * 32 * 16;   // 221184
  const int totalM = NTM * KT * 32 * 16;   // 73728
  int tid = blockIdx.x * blockDim.x + threadIdx.x;
  if (tid < totalG) {
    int e    = tid & 15;
    int lane = (tid >> 4) & 31;
    int kt   = (tid >> 9) % KT;
    int nt   = tid / (KT * 512);
    int n    = nt * 16 + (lane & 15);               // 0..767
    int k    = kt * 32 + ((lane >> 4) << 4) + e;    // 0..287
    float v = 0.f;
    if (k < Hn)           v = w_hh[n * Hn + k];
    else if (k < Hn + Pn) v = w_ih[n * Pn + (k - Hn)];
    Wg[tid] = f2bf(v);
  } else if (tid < totalG + totalM) {
    int t2   = tid - totalG;
    int e    = t2 & 15;
    int lane = (t2 >> 4) & 31;
    int kt   = (t2 >> 9) % KT;
    int nt   = t2 / (KT * 512);
    int n    = nt * 16 + (lane & 15);               // 0..255
    int k    = kt * 32 + ((lane >> 4) << 4) + e;    // 0..287
    float v = 0.f;
    if (k < Hn + Pn - 1)  v = w1[n * (Hn + Pn - 1) + k];
    Wm[t2] = f2bf(v);
  }
}

// ---- fragment loaders ----
__device__ __forceinline__ v16bf load_ah(const unsigned short hb[16][264],
                                         int mrow, int hi, int kt) {
  // A 16x32 in contiguous k order (matches B frags): lanes 0-15 hold K = kt*32+0..15 contiguous,
  // lanes 16-31 hold K = kt*32+16..31.
  BFrag f;
  int kb = kt * 32 + hi * 16;
  f.q[0] = *(const uint4*)&hb[mrow][kb];
  f.q[1] = *(const uint4*)&hb[mrow][kb + 8];
  return f.v;
}
__device__ __forceinline__ v16bf load_bf(const unsigned short* __restrict__ W,
                                         int frag, int lane) {
  const uint4* p = (const uint4*)(W + ((size_t)frag << 9) + (lane << 4));
  BFrag f;
  f.q[0] = p[0];
  f.q[1] = p[1];
  return f.v;
}
__device__ __forceinline__ v16bf make_ax(const float* xv, int hi) {
  // k-tile 8 of A: only K=256..263 (the 8 fused inputs) nonzero -> lanes 0-15, elems 0-7.
  BFrag f;
  f.q[0] = make_uint4(0u, 0u, 0u, 0u);
  f.q[1] = make_uint4(0u, 0u, 0u, 0u);
  if (hi == 0) {
#pragma unroll
    for (int p = 0; p < 8; ++p) f.s[p] = f2bf(xv[p]);
  }
  return f.v;
}
__device__ __forceinline__ v8f wmma_bf(v16bf a, v16bf b, v8f c) {
  return __builtin_amdgcn_wmma_f32_16x16x32_bf16(false, a, false, b, (short)0, c,
                                                 false, false);
}

// ---------------------------------------------------------------------------
// Persistent GRU forecaster: 1 wave (32 threads) owns 16 batch rows for all
// 240 steps; h lives in LDS; weights stream from L2 as pre-swizzled fragments.
// Gate loops are kept as real loops (unroll disabled) so the weight-fragment
// loads are NOT loop-invariant -> no LICM hoist -> no scratch spilling.
// ---------------------------------------------------------------------------
__global__ void __launch_bounds__(32) gru_forecast(
    const float* __restrict__ x,      // (4096, 240, 8)
    const float* __restrict__ b_ih,   // (768)
    const float* __restrict__ b_hh,   // (768)
    const float* __restrict__ b1,     // (256)
    const float* __restrict__ w2,     // (256)
    const float* __restrict__ b2,     // (1)
    const unsigned short* __restrict__ Wg,   // GRU combined B frags
    const unsigned short* __restrict__ Wm,   // MLP  combined B frags
    float* __restrict__ out)          // (4096, 96)
{
  __shared__ unsigned short hbuf[16][264];   // h tile, bf16, padded row stride
  __shared__ unsigned short actT[256][16];   // MLP activations, transposed [u][m]
  __shared__ float dvals[16];
  __shared__ float w2s[256];
  __shared__ float brs[256], bzs[256], bnis[256], bnhs[256], b1s[256];

  const int lane = threadIdx.x;
  const int mrow = lane & 15;   // A-frag row / C-tile column index
  const int hi   = lane >> 4;   // K-half (A/B frags) and M-half (C tiles)
  const int r0   = blockIdx.x << 4;

  for (int i = lane; i < 16 * 264; i += 32) ((unsigned short*)hbuf)[i] = 0;
  for (int i = lane; i < 256; i += 32) {
    brs[i]  = b_ih[i]       + b_hh[i];
    bzs[i]  = b_ih[256 + i] + b_hh[256 + i];
    bnis[i] = b_ih[512 + i];
    bnhs[i] = b_hh[512 + i];
    b1s[i]  = b1[i];
    w2s[i]  = w2[i];
  }
  const float b2v = b2[0];
  const v8f vzero = {0.f, 0.f, 0.f, 0.f, 0.f, 0.f, 0.f, 0.f};
  __syncthreads();

#pragma clang loop unroll(disable)
  for (int t = 0; t < SEQn; ++t) {
    // A fragments for the hidden part (K = 0..255) of this step
    v16bf ah[8];
#pragma unroll
    for (int kt = 0; kt < 8; ++kt) ah[kt] = load_ah(hbuf, mrow, hi, kt);

    float xin[8];
#pragma unroll
    for (int p = 0; p < 8; ++p) xin[p] = 0.f;

    if (t < PREn) {
      // encoder: GRU input = x[:, t, :8]
      const float* xp = x + ((size_t)(r0 + mrow) * SEQn + t) * Pn;
#pragma unroll
      for (int p = 0; p < 8; ++p) xin[p] = xp[p];
    } else {
      // decoder step di: MLP head first (uses pre-update h), then GRU input [xe, d]
      const int di = t - PREn;
      const float* xp = x + ((size_t)(r0 + mrow) * SEQn + (PREn - 1 + di)) * Pn;
      float xe[8];
#pragma unroll
      for (int p = 0; p < 7; ++p) xe[p] = xp[p];
      xe[7] = 0.f;
      v16bf axm = make_ax(xe, hi);

#pragma clang loop unroll(disable)
      for (int i = 0; i < 16; ++i) {
        v8f acc = vzero;
#pragma unroll
        for (int kt = 0; kt < 8; ++kt)
          acc = wmma_bf(ah[kt], load_bf(Wm, i * KT + kt, lane), acc);
        acc = wmma_bf(axm, load_bf(Wm, i * KT + 8, lane), acc);
        float bb = b1s[i * 16 + mrow];
        union { uint4 q; unsigned short s[8]; } tq;
#pragma unroll
        for (int v = 0; v < 8; ++v) {
          float a = acc[v] + bb;
          tq.s[v] = f2bf(a > 0.f ? a : 0.f);         // ReLU
        }
        // C tile (v,l) -> row m = v + 8*hi, col u = i*16 + mrow; store transposed
        *(uint4*)&actT[i * 16 + mrow][hi * 8] = tq.q;
      }
      __syncthreads();
      if (hi == 0) {                                  // w2 dot + sigmoid per row
        float s = b2v;
        for (int u = 0; u < 256; ++u) s += bf2f(actT[u][mrow]) * w2s[u];
        dvals[mrow] = sigf(s);
      }
      __syncthreads();
      float d = dvals[mrow];
#pragma unroll
      for (int p = 0; p < 7; ++p) xin[p] = xe[p];
      xin[7] = d;
      if (di >= TDEC - OUTn && hi == 0)
        out[(size_t)(r0 + mrow) * OUTn + (di - (TDEC - OUTn))] = d;
    }

    v16bf ax = make_ax(xin, hi);   // k-tile 8 A fragment (fused GRU input)

    // ---- all three gates for column tile i computed together, then h update.
    // r tile i pairs elementwise with n tile i (same (vgpr,lane) -> same (m,u)),
    // so no cross-iteration gate storage is needed.
#pragma clang loop unroll(disable)
    for (int i = 0; i < 16; ++i) {
      v8f racc = vzero, zacc = vzero, ghn = vzero;
#pragma unroll
      for (int kt = 0; kt < 8; ++kt) {
        racc = wmma_bf(ah[kt], load_bf(Wg, i * KT + kt, lane), racc);
        zacc = wmma_bf(ah[kt], load_bf(Wg, (16 + i) * KT + kt, lane), zacc);
        ghn  = wmma_bf(ah[kt], load_bf(Wg, (32 + i) * KT + kt, lane), ghn);
      }
      racc = wmma_bf(ax, load_bf(Wg, i * KT + 8, lane), racc);
      zacc = wmma_bf(ax, load_bf(Wg, (16 + i) * KT + 8, lane), zacc);
      v8f gin = wmma_bf(ax, load_bf(Wg, (32 + i) * KT + 8, lane), vzero);

      const int u  = i * 16 + mrow;
      const float br = brs[u], bz = bzs[u], bi = bnis[u], bh = bnhs[u];
#pragma unroll
      for (int v = 0; v < 8; ++v) {
        int m = v + hi * 8;
        float rg = sigf(racc[v] + br);
        float zg = sigf(zacc[v] + bz);
        float ng = tanhfast(gin[v] + bi + rg * (ghn[v] + bh));
        float h0 = bf2f(hbuf[m][u]);
        hbuf[m][u] = f2bf((1.f - zg) * ng + zg * h0);
      }
    }
    __syncthreads();
  }
}

extern "C" void kernel_launch(void* const* d_in, const int* in_sizes, int n_in,
                              void* d_out, int out_size, void* d_ws, size_t ws_size,
                              hipStream_t stream) {
  (void)in_sizes; (void)n_in; (void)out_size; (void)ws_size;
  const float* x    = (const float*)d_in[0];
  const float* w_ih = (const float*)d_in[1];
  const float* w_hh = (const float*)d_in[2];
  const float* b_ih = (const float*)d_in[3];
  const float* b_hh = (const float*)d_in[4];
  const float* w1   = (const float*)d_in[5];
  const float* b1   = (const float*)d_in[6];
  const float* w2   = (const float*)d_in[7];
  const float* b2   = (const float*)d_in[8];

  unsigned short* Wg = (unsigned short*)d_ws;         // 221184 elems (432 KB)
  unsigned short* Wm = Wg + (size_t)NTG * KT * 512;   // 73728 elems (144 KB)
  float* out = (float*)d_out;

  const int totalPrep = (NTG + NTM) * KT * 512;       // 294912
  prep_weights<<<(totalPrep + 255) / 256, 256, 0, stream>>>(w_ih, w_hh, w1, Wg, Wm);

  // 4096 rows / 16 rows-per-wave = 256 independent persistent blocks
  gru_forecast<<<256, 32, 0, stream>>>(x, b_ih, b_hh, b1, w2, b2, Wg, Wm, out);
}
